// CustomMultiheadAttention_31473520345414
// MI455X (gfx1250) — matrix operand order repaired and checked by reading, not compiled, not run on hardware
//
#include <hip/hip_runtime.h>

typedef _Float16 v16h __attribute__((ext_vector_type(16)));
typedef _Float16 v8h  __attribute__((ext_vector_type(8)));
typedef float    v8f  __attribute__((ext_vector_type(8)));

#define B_  2
#define T_  4096
#define C_  512
#define H_  8
#define DK_ 64
#define N3_ 1536
#define BT_ (B_*T_)

union AF { v16h v; v8h h[2]; };

static __device__ __forceinline__ v8f wmma_f16(v16h a, v16h b, v8f c) {
  // D = A(16x32 f16) * B(32x16 f16) + C(16x16 f32)
  return __builtin_amdgcn_wmma_f32_16x16x32_f16(false, a, false, b, (short)0, c,
                                                false, false);
}

static __device__ __forceinline__ v8f vzero8f() {
  v8f z;
  #pragma unroll
  for (int i = 0; i < 8; ++i) z[i] = 0.0f;
  return z;
}

// ---------------- elementwise f32 -> f16 ----------------
__global__ void k_f32_to_f16(const float* __restrict__ src,
                             _Float16* __restrict__ dst, int n) {
  int i = blockIdx.x * blockDim.x + threadIdx.x;
  if (i < n) dst[i] = (_Float16)src[i];
}

// ---------------- transpose f32[rows][cols] -> f16[cols][rows] ----------------
__global__ void k_transpose_f16(const float* __restrict__ src,
                                _Float16* __restrict__ dst, int rows, int cols) {
  int i = blockIdx.x * blockDim.x + threadIdx.x;
  if (i < rows * cols) {
    int n = i / rows;   // output row (= source column)
    int k = i % rows;
    dst[n * rows + k] = (_Float16)src[k * cols + n];
  }
}

// ---------------- QKV projection GEMM ----------------
// xh [BT,512] f16 * WqkvT (n-major [1536][512]) -> Q/K head-major, V transposed.
__global__ void __launch_bounds__(256) k_qkv_gemm(
    const _Float16* __restrict__ xh, const _Float16* __restrict__ wT,
    const float* __restrict__ bias,
    _Float16* __restrict__ Qh, _Float16* __restrict__ Kh,
    _Float16* __restrict__ Vt) {
  const int lane = threadIdx.x & 31;
  const int wave = threadIdx.x >> 5;
  const int hi   = lane >> 4;     // 0 or 1 (wave32 half)
  const int l16  = lane & 15;
  const int wg   = blockIdx.x * 8 + wave;       // 12288 wave-tiles
  const int mt   = wg / 24;                      // 512 M tiles of 16
  const int nt   = wg % 24;                      // 24 N tiles of 64
  const int mbase = mt * 16, nbase = nt * 64;

  v8f acc[4];
  #pragma unroll
  for (int jt = 0; jt < 4; ++jt) acc[jt] = vzero8f();

  for (int kk = 0; kk < 16; ++kk) {             // K = 512, step 32
    AF a;
    const _Float16* ap = xh + (size_t)(mbase + l16) * 512 + kk * 32 + hi * 16;
    a.h[0] = *(const v8h*)(ap);
    a.h[1] = *(const v8h*)(ap + 8);
    #pragma unroll
    for (int jt = 0; jt < 4; ++jt) {
      const _Float16* bp =
          wT + (size_t)(nbase + jt * 16 + l16) * 512 + kk * 32 + hi * 16;
      v16h bm = *(const v16h*)bp;
      acc[jt] = wmma_f16(a.v, bm, acc[jt]);
    }
  }

  #pragma unroll
  for (int jt = 0; jt < 4; ++jt) {
    int col = nbase + jt * 16 + l16;
    float bv = bias[col];
    #pragma unroll
    for (int r = 0; r < 8; ++r) {
      int row = mbase + r + hi * 8;             // [0, 8192)
      int b = row >> 12, t = row & 4095;
      _Float16 val = (_Float16)(acc[jt][r] + bv);
      if (col < 512) {
        int h = col >> 6, dk = col & 63;
        Qh[(size_t)((b * 8 + h) * 4096 + t) * 64 + dk] = val;
      } else if (col < 1024) {
        int c2 = col - 512, h = c2 >> 6, dk = c2 & 63;
        Kh[(size_t)((b * 8 + h) * 4096 + t) * 64 + dk] = val;
      } else {
        int c2 = col - 1024, h = c2 >> 6, dk = c2 & 63;
        Vt[(size_t)((b * 8 + h) * 64 + dk) * 4096 + t] = val;   // transposed
      }
    }
  }
}

// ---------------- flash attention ----------------
// grid = B*H*(T/128) = 512 blocks, 256 threads = 8 independent waves,
// each wave owns a 16-row Q tile, streams 64-wide K/V tiles.
__global__ void __launch_bounds__(256) k_attn(
    const _Float16* __restrict__ Qh, const _Float16* __restrict__ Kh,
    const _Float16* __restrict__ Vt, const float* __restrict__ mask,
    _Float16* __restrict__ Oh) {
  __shared__ __align__(16) _Float16 Plds[8][16][64];   // per-wave P staging
  const int lane = threadIdx.x & 31;
  const int wave = threadIdx.x >> 5;
  const int hi   = lane >> 4;
  const int l16  = lane & 15;
  const int bh   = blockIdx.x >> 5;                    // [0,16)
  const int qblk = blockIdx.x & 31;                    // [0,32)
  const int b = bh >> 3, h = bh & 7;
  const int qrow0 = qblk * 128 + wave * 16;            // abs q row in T
  const float scale = 0.125f;                          // 1/sqrt(64)

  // Q A-fragments (held in registers for the whole kernel)
  AF qa[2];
  {
    const _Float16* qp = Qh + ((size_t)bh * 4096 + qrow0 + l16) * 64;
    #pragma unroll
    for (int ks = 0; ks < 2; ++ks) {
      qa[ks].h[0] = *(const v8h*)(qp + ks * 32 + hi * 16);
      qa[ks].h[1] = *(const v8h*)(qp + ks * 32 + hi * 16 + 8);
    }
  }

  v8f o[4];
  float mstat[8], lstat[8];
  #pragma unroll
  for (int ot = 0; ot < 4; ++ot) o[ot] = vzero8f();
  #pragma unroll
  for (int r = 0; r < 8; ++r) { mstat[r] = -1e30f; lstat[r] = 0.0f; }

  const _Float16* kbase = Kh + (size_t)bh * 4096 * 64;
  const _Float16* vbase = Vt + (size_t)bh * 64 * 4096;

  for (int kt = 0; kt < 64; ++kt) {
    const int kcol0 = kt * 64;

    // S = Q * K^T  (4 column tiles of 16)
    v8f s[4];
    #pragma unroll
    for (int jt = 0; jt < 4; ++jt) {
      const _Float16* kp =
          kbase + (size_t)(kcol0 + jt * 16 + l16) * 64 + hi * 16;
      v16h b0 = *(const v16h*)(kp);
      v16h b1 = *(const v16h*)(kp + 32);
      v8f a = vzero8f();
      a = wmma_f16(qa[0].v, b0, a);
      a = wmma_f16(qa[1].v, b1, a);
      s[jt] = a;
    }

    if (kt + 1 < 64)   // prefetch next K tile (global_prefetch_b8)
      __builtin_prefetch(kbase + (size_t)(kcol0 + 64 + (lane << 1)) * 64, 0, 1);

    // scale + mask + per-lane row max
    float rmax[8];
    #pragma unroll
    for (int r = 0; r < 8; ++r) {
      const int qrow = qrow0 + r + hi * 8;
      float mrow = -1e30f;
      #pragma unroll
      for (int jt = 0; jt < 4; ++jt) {
        const int col = kcol0 + jt * 16 + l16;
        float v = s[jt][r] * scale + mask[(size_t)qrow * 4096 + col];
        s[jt][r] = v;
        mrow = fmaxf(mrow, v);
      }
      rmax[r] = mrow;
    }
    // cross-lane row reductions stay inside each 16-lane half (C layout)
    #pragma unroll
    for (int r = 0; r < 8; ++r) {
      float v = rmax[r];
      v = fmaxf(v, __shfl_xor(v, 1, 32));
      v = fmaxf(v, __shfl_xor(v, 2, 32));
      v = fmaxf(v, __shfl_xor(v, 4, 32));
      v = fmaxf(v, __shfl_xor(v, 8, 32));
      rmax[r] = v;
    }
    #pragma unroll
    for (int r = 0; r < 8; ++r) {
      const float mnew  = fmaxf(mstat[r], rmax[r]);
      const float alpha = __expf(mstat[r] - mnew);
      mstat[r] = mnew;
      float rsum = 0.0f;
      #pragma unroll
      for (int jt = 0; jt < 4; ++jt) {
        float p = __expf(s[jt][r] - mnew);
        s[jt][r] = p;
        rsum += p;
      }
      rsum += __shfl_xor(rsum, 1, 32);
      rsum += __shfl_xor(rsum, 2, 32);
      rsum += __shfl_xor(rsum, 4, 32);
      rsum += __shfl_xor(rsum, 8, 32);
      lstat[r] = lstat[r] * alpha + rsum;
      #pragma unroll
      for (int ot = 0; ot < 4; ++ot) o[ot][r] *= alpha;
    }

    // P: C-layout regs -> LDS -> A-layout fragments
    #pragma unroll
    for (int jt = 0; jt < 4; ++jt)
      #pragma unroll
      for (int r = 0; r < 8; ++r)
        Plds[wave][r + hi * 8][jt * 16 + l16] = (_Float16)s[jt][r];
    __syncthreads();
    AF pa[2];
    #pragma unroll
    for (int ks = 0; ks < 2; ++ks) {
      const _Float16* pp = &Plds[wave][l16][ks * 32 + hi * 16];
      pa[ks].h[0] = *(const v8h*)(pp);
      pa[ks].h[1] = *(const v8h*)(pp + 8);
    }
    __syncthreads();

    // O += P * V   (V stored k-major -> contiguous B fragments)
    #pragma unroll
    for (int ot = 0; ot < 4; ++ot) {
      const _Float16* vp =
          vbase + (size_t)(ot * 16 + l16) * 4096 + kcol0 + hi * 16;
      v16h b0 = *(const v16h*)(vp);
      v16h b1 = *(const v16h*)(vp + 32);
      o[ot] = wmma_f16(pa[0].v, b0, o[ot]);
      o[ot] = wmma_f16(pa[1].v, b1, o[ot]);
    }
  }

  // normalize and store O as f16, channel c = h*64 + n
  #pragma unroll
  for (int r = 0; r < 8; ++r) {
    const float inv = 1.0f / lstat[r];
    const int t = qrow0 + r + hi * 8;
    const size_t orow = ((size_t)b * 4096 + t) * 512 + h * 64;
    #pragma unroll
    for (int ot = 0; ot < 4; ++ot)
      Oh[orow + ot * 16 + l16] = (_Float16)(o[ot][r] * inv);
  }
}

// ---------------- output projection GEMM ----------------
__global__ void __launch_bounds__(256) k_out_gemm(
    const _Float16* __restrict__ Oh, const _Float16* __restrict__ wT,
    const float* __restrict__ bias, float* __restrict__ out) {
  const int lane = threadIdx.x & 31;
  const int wave = threadIdx.x >> 5;
  const int hi   = lane >> 4;
  const int l16  = lane & 15;
  const int wg   = blockIdx.x * 8 + wave;       // 4096 wave-tiles
  const int mt   = wg >> 3;                     // 512 M tiles
  const int nt   = wg & 7;                      // 8 N tiles of 64
  const int mbase = mt * 16, nbase = nt * 64;

  v8f acc[4];
  #pragma unroll
  for (int jt = 0; jt < 4; ++jt) acc[jt] = vzero8f();

  for (int kk = 0; kk < 16; ++kk) {
    AF a;
    const _Float16* ap = Oh + (size_t)(mbase + l16) * 512 + kk * 32 + hi * 16;
    a.h[0] = *(const v8h*)(ap);
    a.h[1] = *(const v8h*)(ap + 8);
    #pragma unroll
    for (int jt = 0; jt < 4; ++jt) {
      const _Float16* bp =
          wT + (size_t)(nbase + jt * 16 + l16) * 512 + kk * 32 + hi * 16;
      v16h bm = *(const v16h*)bp;
      acc[jt] = wmma_f16(a.v, bm, acc[jt]);
    }
  }

  #pragma unroll
  for (int jt = 0; jt < 4; ++jt) {
    const int col = nbase + jt * 16 + l16;
    const float bv = bias[col];
    #pragma unroll
    for (int r = 0; r < 8; ++r) {
      const int row = mbase + r + hi * 8;
      out[(size_t)row * 512 + col] = acc[jt][r] + bv;
    }
  }
}

extern "C" void kernel_launch(void* const* d_in, const int* in_sizes, int n_in,
                              void* d_out, int out_size, void* d_ws,
                              size_t ws_size, hipStream_t stream) {
  const float* x    = (const float*)d_in[0];   // [2,4096,512]
  const float* mask = (const float*)d_in[1];   // [4096,4096]
  const float* Wqkv = (const float*)d_in[2];   // [512,1536]
  const float* bqkv = (const float*)d_in[3];   // [1536]
  const float* Wout = (const float*)d_in[4];   // [512,512]
  const float* bout = (const float*)d_in[5];   // [512]
  float* out = (float*)d_out;

  char* ws = (char*)d_ws;
  _Float16* xh    = (_Float16*)(ws);                          //  8 MiB
  _Float16* Qh    = (_Float16*)(ws + (size_t)8  * (1 << 20)); //  8 MiB
  _Float16* Kh    = (_Float16*)(ws + (size_t)16 * (1 << 20)); //  8 MiB
  _Float16* Vt    = (_Float16*)(ws + (size_t)24 * (1 << 20)); //  8 MiB
  _Float16* Oh    = (_Float16*)(ws + (size_t)32 * (1 << 20)); //  8 MiB
  _Float16* WqkvT = (_Float16*)(ws + (size_t)40 * (1 << 20)); // 1.5 MiB
  _Float16* WoutT = (_Float16*)(ws + (size_t)42 * (1 << 20)); // 0.5 MiB

  k_f32_to_f16<<<(BT_ * C_) / 256, 256, 0, stream>>>(x, xh, BT_ * C_);
  k_transpose_f16<<<(C_ * N3_) / 256, 256, 0, stream>>>(Wqkv, WqkvT, C_, N3_);
  k_transpose_f16<<<(C_ * C_) / 256, 256, 0, stream>>>(Wout, WoutT, C_, C_);
  k_qkv_gemm<<<1536, 256, 0, stream>>>(xh, WqkvT, bqkv, Qh, Kh, Vt);
  k_attn<<<512, 256, 0, stream>>>(Qh, Kh, Vt, mask, Oh);
  k_out_gemm<<<512, 256, 0, stream>>>(Oh, WoutT, bout, out);
}
